// MultiheadSelfAttention_72000831750501
// MI455X (gfx1250) — matrix operand order repaired and checked by reading, not compiled, not run on hardware
//
#include <hip/hip_runtime.h>
#include <hip/hip_bf16.h>

// ---------------- WMMA types & helpers (gfx1250, wave32) ----------------
typedef __attribute__((ext_vector_type(16))) __bf16 v16bf;
typedef __attribute__((ext_vector_type(8)))  float  v8f;

__device__ inline v8f wmma_bf16(v16bf a, v16bf b, v8f c) {
    // D = A(16x32 bf16) * B(32x16 bf16) + C(16x16 f32)
    return __builtin_amdgcn_wmma_f32_16x16x32_bf16(
        /*neg_a=*/false, a, /*neg_b=*/false, b,
        /*c_mod=*/(short)0, c, /*reuse_a=*/false, /*reuse_b=*/false);
}

// CDNA5 async DMA: global -> LDS, tracked by ASYNCcnt (no VGPR data path).
// Low 32 bits of a generic pointer to __shared__ are the LDS byte address.
__device__ inline void async_load_lds_b128(void* lds_dst, const void* gsrc) {
    unsigned int loff = (unsigned int)(unsigned long long)lds_dst;
    asm volatile("global_load_async_to_lds_b128 %0, %1, off"
                 :: "v"(loff), "v"(gsrc)
                 : "memory");
}
__device__ inline void wait_async0() {
    asm volatile("s_wait_asynccnt 0" ::: "memory");
}

// A-matrix fragment (16x32 bf16), source is row-major [16][ld] in LDS.
// Lanes 0-15: M=lane, K in {0..15}; lanes 16-31: M=lane-16, K in {16..31}.
__device__ inline v16bf frag_a(const __bf16* p0, int ld) {
    int lane = threadIdx.x & 31;
    int m = lane & 15, hs = lane >> 4;
    const __bf16* p = p0 + m * ld;
    v16bf a;
#pragma unroll
    for (int i = 0; i < 4; ++i) {
        int k = 2 * i + hs * 16;
        a[2 * i]     = p[k];
        a[2 * i + 1] = p[k + 1];
        a[8 + 2 * i] = p[8 + k];
        a[9 + 2 * i] = p[9 + k];
    }
    return a;
}

// B-matrix fragment (32x16) with B(k,n) = p0[n*ld + k] (transposed source).
__device__ inline v16bf frag_bt(const __bf16* p0, int ld) {
    int lane = threadIdx.x & 31;
    int n = lane & 15, hs = lane >> 4;
    const __bf16* p = p0 + n * ld + hs * 16;
    v16bf b;
#pragma unroll
    for (int i = 0; i < 16; ++i) b[i] = p[i];
    return b;
}

// B-matrix fragment (32x16) with B(k,n) = p0[k*ld + n] (row-major source).
__device__ inline v16bf frag_b(const __bf16* p0, int ld) {
    int lane = threadIdx.x & 31;
    int n = lane & 15, hs = lane >> 4;
    const __bf16* p = p0 + hs * 16 * ld + n;
    v16bf b;
#pragma unroll
    for (int i = 0; i < 16; ++i) b[i] = p[i * ld];
    return b;
}

// ---------------- problem constants ----------------
#define S_LEN  2048
#define DMODEL 1024
#define HEADS  16
#define DKH    64
#define BM     128
#define BN     64

// ---------------- fp32 -> bf16 conversion ----------------
__global__ void f32_to_bf16(const float* __restrict__ src, __bf16* __restrict__ dst, int n) {
    int i4 = (blockIdx.x * blockDim.x + threadIdx.x) * 4;
    if (i4 + 3 < n) {
        float4 v = *(const float4*)&src[i4];
        dst[i4 + 0] = (__bf16)v.x;
        dst[i4 + 1] = (__bf16)v.y;
        dst[i4 + 2] = (__bf16)v.z;
        dst[i4 + 3] = (__bf16)v.w;
    }
}

// ------- C = A * W^T  (A:[M][Kd], W:[N][Kd] bf16; C bf16 or f32) -------
// 128x128 block tile, 8 waves in a 4(M) x 2(N) grid; each wave: 32x64 via
// 2 A-frags x 4 B-frags = 8 WMMA per 32-deep K chunk. Double-buffered LDS
// fed by async global->LDS DMA.
__global__ __launch_bounds__(256) void gemm_bf16_nt(
    const __bf16* __restrict__ A, const __bf16* __restrict__ W,
    __bf16* __restrict__ Cb, float* __restrict__ Cf,
    int M, int N, int Kd)
{
    __shared__ __align__(16) __bf16 As[2][128 * 32];
    __shared__ __align__(16) __bf16 Ws[2][128 * 32];

    const int bm   = blockIdx.x * 128;
    const int bn   = blockIdx.y * 128;
    const int wave = threadIdx.x >> 5;
    const int lane = threadIdx.x & 31;
    const int mw   = wave & 3;          // 0..3  -> M offset mw*32
    const int nw   = wave >> 2;         // 0..1  -> N offset nw*64
    const int hs   = lane >> 4;
    const int nn   = lane & 15;

    const v8f z = {0.f, 0.f, 0.f, 0.f, 0.f, 0.f, 0.f, 0.f};
    v8f acc[2][4];
#pragma unroll
    for (int i = 0; i < 2; ++i)
#pragma unroll
        for (int j = 0; j < 4; ++j) acc[i][j] = z;

    // prologue: async-stage K-chunk 0 into buffer 0
#pragma unroll
    for (int t = 0; t < 2; ++t) {
        int idx = (threadIdx.x + t * 256) * 8;      // 0..4095 bf16 elements
        int r = idx >> 5, c = idx & 31;
        async_load_lds_b128(&As[0][idx], &A[(size_t)(bm + r) * Kd + c]);
        async_load_lds_b128(&Ws[0][idx], &W[(size_t)(bn + r) * Kd + c]);
    }

    for (int k0 = 0, it = 0; k0 < Kd; k0 += 32, ++it) {
        const int cur = it & 1;
        wait_async0();                 // our async loads into `cur` complete
        __syncthreads();               // all waves' loads visible, prev reads done

        if (k0 + 32 < Kd) {            // overlap DMA of next chunk with WMMA
            const int nxt = cur ^ 1;
#pragma unroll
            for (int t = 0; t < 2; ++t) {
                int idx = (threadIdx.x + t * 256) * 8;
                int r = idx >> 5, c = idx & 31;
                async_load_lds_b128(&As[nxt][idx],
                                    &A[(size_t)(bm + r) * Kd + (k0 + 32) + c]);
                async_load_lds_b128(&Ws[nxt][idx],
                                    &W[(size_t)(bn + r) * Kd + (k0 + 32) + c]);
            }
        }

        v16bf a0 = frag_a(&As[cur][(mw * 32) * 32], 32);
        v16bf a1 = frag_a(&As[cur][(mw * 32 + 16) * 32], 32);
#pragma unroll
        for (int nt = 0; nt < 4; ++nt) {
            v16bf b = frag_bt(&Ws[cur][(nw * 64 + nt * 16) * 32], 32);
            acc[0][nt] = wmma_bf16(a0, b, acc[0][nt]);
            acc[1][nt] = wmma_bf16(a1, b, acc[1][nt]);
        }
    }

    // epilogue: C lane layout — VGPR r, lanes<16 => M=r, lanes>=16 => M=r+8
#pragma unroll
    for (int mt = 0; mt < 2; ++mt)
#pragma unroll
        for (int nt = 0; nt < 4; ++nt)
#pragma unroll
            for (int r = 0; r < 8; ++r) {
                size_t row = (size_t)(bm + mw * 32 + mt * 16 + r + hs * 8);
                size_t col = (size_t)(bn + nw * 64 + nt * 16 + nn);
                float v = acc[mt][nt][r];
                if (Cb) Cb[row * N + col] = (__bf16)v;
                else    Cf[row * N + col] = v;
            }
}

// ---------------- RoPE in-place on bf16 Q and K ----------------
__global__ void rope_kernel(__bf16* __restrict__ Q, __bf16* __restrict__ K, int npairs) {
    int idx = blockIdx.x * blockDim.x + threadIdx.x;
    if (idx >= npairs) return;
    int pr  = idx % (DMODEL / 2);         // pair index within row; col = 2*pr
    int row = idx / (DMODEL / 2);         // b*S + s
    int pos = row % S_LEN;
    int i   = pr & (DKH / 2 - 1);         // pair index within head
    float freq = __powf(10000.0f, -(float)(2 * i) / (float)DKH);
    float phi  = (float)pos * freq;
    float sn, cs;
    __sincosf(phi, &sn, &cs);
    size_t e0 = (size_t)row * DMODEL + 2 * pr;
    float qe = (float)Q[e0], qo = (float)Q[e0 + 1];
    Q[e0]     = (__bf16)(qe * cs - qo * sn);
    Q[e0 + 1] = (__bf16)(qe * sn + qo * cs);
    float ke = (float)K[e0], ko = (float)K[e0 + 1];
    K[e0]     = (__bf16)(ke * cs - ko * sn);
    K[e0 + 1] = (__bf16)(ke * sn + ko * cs);
}

// ---------------- flash attention (causal, online softmax) ----------------
// grid: (S/BM, HEADS, B); block: 256 threads = 8 waves; wave w owns 16 query
// rows. KV tiles double-buffered in LDS, staged by async DMA.
__global__ __launch_bounds__(256) void flash_attn(
    const __bf16* __restrict__ Q, const __bf16* __restrict__ Km,
    const __bf16* __restrict__ V, __bf16* __restrict__ O)
{
    __shared__ __align__(16) __bf16 Qs[BM * DKH];          // 16 KB
    __shared__ __align__(16) __bf16 Ks[2][BN * DKH];       // 16 KB
    __shared__ __align__(16) __bf16 Vs[2][BN * DKH];       // 16 KB
    __shared__ __align__(16) __bf16 Ps[8][16 * BN];        // 16 KB (wave-private P)

    const int q0   = blockIdx.x * BM;
    const int h    = blockIdx.y;
    const int b    = blockIdx.z;
    const int wave = threadIdx.x >> 5;
    const int lane = threadIdx.x & 31;
    const int hs   = lane >> 4;
    const int nn   = lane & 15;
    const size_t rowbase = (size_t)b * S_LEN * DMODEL + (size_t)h * DKH;

    // async-stage KV tile j=0 (BN*DKH = 4096 elems = 2 rounds of 256x8)
#pragma unroll
    for (int t = 0; t < 2; ++t) {
        int idx = (threadIdx.x + t * 256) * 8;
        int r = idx / DKH, c = idx % DKH;
        async_load_lds_b128(&Ks[0][idx], &Km[rowbase + (size_t)r * DMODEL + c]);
        async_load_lds_b128(&Vs[0][idx], &V [rowbase + (size_t)r * DMODEL + c]);
    }
    // async-stage the 128x64 query tile (4 rounds)
#pragma unroll
    for (int t = 0; t < 4; ++t) {
        int idx = (threadIdx.x + t * 256) * 8;
        int r = idx / DKH, c = idx % DKH;
        async_load_lds_b128(&Qs[idx], &Q[rowbase + (size_t)(q0 + r) * DMODEL + c]);
    }
    wait_async0();
    __syncthreads();

    v16bf qa0 = frag_a(Qs + wave * 16 * DKH, DKH);        // dk 0..31
    v16bf qa1 = frag_a(Qs + wave * 16 * DKH + 32, DKH);   // dk 32..63

    const v8f z = {0.f, 0.f, 0.f, 0.f, 0.f, 0.f, 0.f, 0.f};
    v8f o[4];
#pragma unroll
    for (int i = 0; i < 4; ++i) o[i] = z;
    float mrow[8], lrow[8];
#pragma unroll
    for (int r = 0; r < 8; ++r) { mrow[r] = -3.0e38f; lrow[r] = 0.f; }

    const int qrow_base = q0 + wave * 16 + hs * 8;        // + r -> query index
    const int jmax = (q0 + BM - 1) / BN;

    for (int j = 0; j <= jmax; ++j) {
        const int cur = j & 1;
        if (j > 0) {
            wait_async0();             // our DMA into `cur` complete
            __syncthreads();           // all waves: data visible, old reads done
        }
        if (j < jmax) {                // overlap next KV tile DMA with compute
            const int nxt = cur ^ 1;
            const int kb2 = (j + 1) * BN;
#pragma unroll
            for (int t = 0; t < 2; ++t) {
                int idx = (threadIdx.x + t * 256) * 8;
                int r = idx / DKH, c = idx % DKH;
                async_load_lds_b128(&Ks[nxt][idx],
                                    &Km[rowbase + (size_t)(kb2 + r) * DMODEL + c]);
                async_load_lds_b128(&Vs[nxt][idx],
                                    &V [rowbase + (size_t)(kb2 + r) * DMODEL + c]);
            }
        }
        const int kbase = j * BN;

        // S = Q * K^T : 4 N-tiles x 2 K-chunks = 8 WMMA
        v8f s[4];
#pragma unroll
        for (int nt = 0; nt < 4; ++nt) s[nt] = z;
#pragma unroll
        for (int nt = 0; nt < 4; ++nt) {
            v16bf b0 = frag_bt(&Ks[cur][nt * 16 * DKH], DKH);
            s[nt] = wmma_bf16(qa0, b0, s[nt]);
            v16bf b1 = frag_bt(&Ks[cur][nt * 16 * DKH + 32], DKH);
            s[nt] = wmma_bf16(qa1, b1, s[nt]);
        }

        // scale (1/sqrt(64)) + causal mask
#pragma unroll
        for (int nt = 0; nt < 4; ++nt) {
            int kidx = kbase + nt * 16 + nn;
#pragma unroll
            for (int r = 0; r < 8; ++r) {
                float val = s[nt][r] * 0.125f;
                s[nt][r] = (kidx <= qrow_base + r) ? val : -3.0e38f;
            }
        }

        // online softmax: row max / rescale / exp / row sum over 16-lane groups
#pragma unroll
        for (int r = 0; r < 8; ++r) {
            float v = fmaxf(fmaxf(s[0][r], s[1][r]), fmaxf(s[2][r], s[3][r]));
            v = fmaxf(v, __shfl_xor(v, 1, 16));
            v = fmaxf(v, __shfl_xor(v, 2, 16));
            v = fmaxf(v, __shfl_xor(v, 4, 16));
            v = fmaxf(v, __shfl_xor(v, 8, 16));
            float mnew = fmaxf(mrow[r], v);
            float corr = __expf(mrow[r] - mnew);
            mrow[r] = mnew;
            lrow[r] *= corr;
#pragma unroll
            for (int nt = 0; nt < 4; ++nt) o[nt][r] *= corr;
            float rsum = 0.f;
#pragma unroll
            for (int nt = 0; nt < 4; ++nt) {
                float p = __expf(s[nt][r] - mnew);
                s[nt][r] = p;
                rsum += p;
            }
            rsum += __shfl_xor(rsum, 1, 16);
            rsum += __shfl_xor(rsum, 2, 16);
            rsum += __shfl_xor(rsum, 4, 16);
            rsum += __shfl_xor(rsum, 8, 16);
            lrow[r] += rsum;
        }

        // spill P (16 x BN bf16) to wave-private LDS, then P*V via WMMA
        __bf16* pw = &Ps[wave][0];
#pragma unroll
        for (int nt = 0; nt < 4; ++nt)
#pragma unroll
            for (int r = 0; r < 8; ++r)
                pw[(r + hs * 8) * BN + nt * 16 + nn] = (__bf16)s[nt][r];

#pragma unroll
        for (int kc = 0; kc < 2; ++kc) {
            v16bf pa = frag_a(pw + kc * 32, BN);
#pragma unroll
            for (int nt = 0; nt < 4; ++nt) {
                v16bf vb = frag_b(&Vs[cur][kc * 32 * DKH + nt * 16], DKH);
                o[nt] = wmma_bf16(pa, vb, o[nt]);
            }
        }
    }

    // epilogue: normalize and write bf16 head output
#pragma unroll
    for (int nt = 0; nt < 4; ++nt)
#pragma unroll
        for (int r = 0; r < 8; ++r) {
            float val = o[nt][r] / lrow[r];
            size_t row = (size_t)(q0 + wave * 16 + r + hs * 8);
            O[rowbase + row * DMODEL + nt * 16 + nn] = (__bf16)val;
        }
}

// ---------------- launch ----------------
extern "C" void kernel_launch(void* const* d_in, const int* in_sizes, int n_in,
                              void* d_out, int out_size, void* d_ws, size_t ws_size,
                              hipStream_t stream) {
    const float* x  = (const float*)d_in[0];
    const float* Wq = (const float*)d_in[1];
    const float* Wk = (const float*)d_in[2];
    const float* Wv = (const float*)d_in[3];
    const float* Wo = (const float*)d_in[4];
    float* out = (float*)d_out;

    const int MROWS = 2 * S_LEN;                      // 4096
    const size_t XN = (size_t)MROWS * DMODEL;         // 4,194,304
    const size_t WN = (size_t)DMODEL * DMODEL;        // 1,048,576

    __bf16* ws  = (__bf16*)d_ws;                      // ~48 MB bf16 working set
    __bf16* Xb  = ws;
    __bf16* Wqb = Xb  + XN;
    __bf16* Wkb = Wqb + WN;
    __bf16* Wvb = Wkb + WN;
    __bf16* Wob = Wvb + WN;
    __bf16* Qb  = Wob + WN;
    __bf16* Kb  = Qb  + XN;
    __bf16* Vb  = Kb  + XN;
    __bf16* Ob  = Vb  + XN;

    f32_to_bf16<<<(int)(XN / 4 / 256), 256, 0, stream>>>(x,  Xb,  (int)XN);
    f32_to_bf16<<<(int)(WN / 4 / 256), 256, 0, stream>>>(Wq, Wqb, (int)WN);
    f32_to_bf16<<<(int)(WN / 4 / 256), 256, 0, stream>>>(Wk, Wkb, (int)WN);
    f32_to_bf16<<<(int)(WN / 4 / 256), 256, 0, stream>>>(Wv, Wvb, (int)WN);
    f32_to_bf16<<<(int)(WN / 4 / 256), 256, 0, stream>>>(Wo, Wob, (int)WN);

    dim3 g(MROWS / 128, DMODEL / 128);                // 32 x 8 blocks
    gemm_bf16_nt<<<g, 256, 0, stream>>>(Xb, Wqb, Qb, nullptr, MROWS, DMODEL, DMODEL);
    gemm_bf16_nt<<<g, 256, 0, stream>>>(Xb, Wkb, Kb, nullptr, MROWS, DMODEL, DMODEL);
    gemm_bf16_nt<<<g, 256, 0, stream>>>(Xb, Wvb, Vb, nullptr, MROWS, DMODEL, DMODEL);

    int npairs = MROWS * (DMODEL / 2);
    rope_kernel<<<(npairs + 255) / 256, 256, 0, stream>>>(Qb, Kb, npairs);

    dim3 ga(S_LEN / BM, HEADS, 2);                    // 16 x 16 x 2 blocks
    flash_attn<<<ga, 256, 0, stream>>>(Qb, Kb, Vb, Ob);

    gemm_bf16_nt<<<g, 256, 0, stream>>>(Ob, Wob, nullptr, out, MROWS, DMODEL, DMODEL);
}
